// DilatedAttention_46213848105592
// MI455X (gfx1250) — matrix operand order repaired and checked by reading, not compiled, not run on hardware
//
#include <hip/hip_runtime.h>
#include <hip/hip_bf16.h>
#include <cstdint>
#include <cstddef>

#define B_  2
#define T_  4096
#define E_  1024
#define H_  16
#define D_  64
#define MROWS (B_ * T_)   // 8192

typedef __bf16          v16bf __attribute__((ext_vector_type(16)));
typedef unsigned short  v16us __attribute__((ext_vector_type(16)));
typedef unsigned int    v8ui  __attribute__((ext_vector_type(8)));
typedef float           v8f   __attribute__((ext_vector_type(8)));

// ---------- helpers ----------
static __device__ __forceinline__ unsigned short f2bf(float f) {
  union { float f; unsigned u; } c; c.f = f;
  unsigned u = c.u;
  u += 0x7fffu + ((u >> 16) & 1u);          // round-to-nearest-even
  return (unsigned short)(u >> 16);
}

static __device__ __forceinline__ v16bf bf_from_us(v16us u) {
  return __builtin_bit_cast(v16bf, u);
}
static __device__ __forceinline__ v16bf bf_from_pairs(v8ui u) {
  return __builtin_bit_cast(v16bf, u);
}

// A-matrix 16x32 bf16 layout: VGPR v holds K pair starting at a_base(v) + 8*half
static __device__ __forceinline__ int a_base(int v) {
  return (v < 4) ? (2 * v) : (16 + 2 * (v - 4));
}

static __device__ __forceinline__ v8f wmma_bf16(v16bf a, v16bf b, v8f c) {
  return __builtin_amdgcn_wmma_f32_16x16x32_bf16(false, a, false, b, (short)0, c,
                                                 false, false);
}

// generic LDS pointer -> LDS byte offset (aperture layout: addr[31:0] = LDS offset)
static __device__ __forceinline__ unsigned lds_off(const void* p) {
  return (unsigned)(uintptr_t)p;
}

// async global -> LDS, 16 bytes per lane (GVS form: SGPR base + 32-bit byte offset)
static __device__ __forceinline__ void async_g2l_b128(const void* gbase,
                                                      unsigned voff_bytes,
                                                      unsigned lds_byte_addr) {
  asm volatile("global_load_async_to_lds_b128 %0, %1, %2"
               :
               : "v"(lds_byte_addr), "v"(voff_bytes),
                 "s"((unsigned long long)(uintptr_t)gbase)
               : "memory");
}
static __device__ __forceinline__ void wait_async0() {
  asm volatile("s_wait_asynccnt 0x0" ::: "memory");
}

// ---------- elementwise kernels ----------
__global__ __launch_bounds__(256) void cvt_bf16_kernel(const float* __restrict__ in,
                                                       unsigned short* __restrict__ out,
                                                       int n) {
  for (int i = blockIdx.x * blockDim.x + threadIdx.x; i < n;
       i += gridDim.x * blockDim.x)
    out[i] = f2bf(in[i]);
}

__global__ __launch_bounds__(256) void fill_kernel(float* __restrict__ p, float val, int n) {
  for (int i = blockIdx.x * blockDim.x + threadIdx.x; i < n;
       i += gridDim.x * blockDim.x)
    p[i] = val;
}

// oacc [B,T,H,64] f32 / zacc [B,T,H] -> A bf16 [B*T, E] (E = h*64+dd)
__global__ __launch_bounds__(256) void merge_kernel(const float* __restrict__ oacc,
                                                    const float* __restrict__ zacc,
                                                    unsigned short* __restrict__ Abf,
                                                    int n) {
  for (int i = blockIdx.x * blockDim.x + threadIdx.x; i < n;
       i += gridDim.x * blockDim.x) {
    int dd = i & 63;
    int midx = i >> 6;               // (b*T+t)*H + h
    int hh = midx & (H_ - 1);
    int bt = midx >> 4;
    float v = oacc[i] / zacc[midx];
    Abf[(size_t)bt * E_ + hh * 64 + dd] = f2bf(v);
  }
}

// ---------- WMMA GEMM: C[M,1024] = A_bf16[M,1024] @ W_bf16[1024,1024] ----------
// Double-buffered LDS, tiles fetched with async global->LDS DMA (ASYNCcnt path).
// mode 0: outF[m*1024+n] = acc + bias[n]            (final projection, f32)
// mode 1: outBf[((b*H+h)*T+t)*64+dd] = bf16((acc+bias[n])*scale)   (QKV, bf16)
__global__ __launch_bounds__(256) void gemm_kernel(const unsigned short* __restrict__ A,
                                                   const unsigned short* __restrict__ W,
                                                   const float* __restrict__ bias,
                                                   float scale, int mode,
                                                   unsigned short* __restrict__ outBf,
                                                   float* __restrict__ outF) {
  constexpr int K_ = E_, N_ = E_;
  constexpr int NK = K_ / 32;
  __shared__ __align__(16) unsigned short As[2][128 * 40]; // stride 40 (80B, 16B-align)
  __shared__ __align__(16) unsigned short Bs[2][32 * 72];  // stride 72 (144B, 16B-align)

  const int tid = threadIdx.x;
  const int lane = tid & 31, wid = tid >> 5;
  const int waveM = wid & 3, waveN = wid >> 2;       // 4 x 2 wave grid
  const int rowBase = blockIdx.y * 128;
  const int colBase = blockIdx.x * 64;
  const int lh = lane >> 4, n16 = lane & 15;

  v8f zero = {0.f, 0.f, 0.f, 0.f, 0.f, 0.f, 0.f, 0.f};
  v8f acc[2][2];
#pragma unroll
  for (int mi = 0; mi < 2; ++mi)
#pragma unroll
    for (int ni = 0; ni < 2; ++ni) acc[mi][ni] = zero;

  auto issueTile = [&](int kt, int buf) {
    const int k0 = kt * 32;
#pragma unroll
    for (int i = 0; i < 2; ++i) {                    // A tile 128x32
      int row = (tid >> 2) + 64 * i, c8 = (tid & 3) * 8;
      unsigned voff = (unsigned)((((size_t)(rowBase + row)) * K_ + k0 + c8) * 2);
      async_g2l_b128(A, voff, lds_off(&As[buf][row * 40 + c8]));
    }
    {                                                // W tile 32x64
      int row = tid >> 3, c8 = (tid & 7) * 8;
      unsigned voff = (unsigned)((((size_t)(k0 + row)) * N_ + colBase + c8) * 2);
      async_g2l_b128(W, voff, lds_off(&Bs[buf][row * 72 + c8]));
    }
  };

  issueTile(0, 0);
  wait_async0();
  __syncthreads();

  for (int kt = 0; kt < NK; ++kt) {
    const int buf = kt & 1;
    if (kt + 1 < NK) issueTile(kt + 1, buf ^ 1);     // DMA next tile during compute

    v16bf afr[2], bfr[2];
#pragma unroll
    for (int mi = 0; mi < 2; ++mi) {
      v8ui u;
      int ro = (waveM * 32 + mi * 16 + n16) * 40;
#pragma unroll
      for (int v = 0; v < 8; ++v)
        u[v] = *(const unsigned int*)&As[buf][ro + a_base(v) + 8 * lh];
      afr[mi] = bf_from_pairs(u);
    }
#pragma unroll
    for (int ni = 0; ni < 2; ++ni) {
      v8ui u;
      int co = waveN * 32 + ni * 16 + n16;
#pragma unroll
      for (int j = 0; j < 8; ++j)
        u[j] = (unsigned)Bs[buf][(a_base(j) + 8 * lh) * 72 + co] |
               ((unsigned)Bs[buf][(a_base(j) + 8 * lh + 1) * 72 + co] << 16);
      bfr[ni] = bf_from_pairs(u);
    }
#pragma unroll
    for (int mi = 0; mi < 2; ++mi)
#pragma unroll
      for (int ni = 0; ni < 2; ++ni)
        acc[mi][ni] = wmma_bf16(afr[mi], bfr[ni], acc[mi][ni]);

    if (kt + 1 < NK) wait_async0();                  // next tile landed
    __syncthreads();                                 // all waves done reading buf
  }

  // epilogue: D layout -> lanes 0-15 (M=v), lanes 16-31 (M=v+8), N = lane&15
#pragma unroll
  for (int mi = 0; mi < 2; ++mi)
#pragma unroll
    for (int ni = 0; ni < 2; ++ni) {
      int gn = colBase + waveN * 32 + ni * 16 + n16;
      float bv = bias[gn];
#pragma unroll
      for (int v = 0; v < 8; ++v) {
        int gm = rowBase + waveM * 32 + mi * 16 + v + 8 * lh;
        float val = (acc[mi][ni][v] + bv) * scale;
        if (mode == 0) {
          outF[(size_t)gm * N_ + gn] = val;
        } else {
          int bb = gm >> 12, tt = gm & (T_ - 1);
          int hh = gn >> 6, dd = gn & 63;
          outBf[(((size_t)(bb * H_ + hh)) * T_ + tt) * D_ + dd] = f2bf(val);
        }
      }
    }
}

// ---------- dilated flash attention, one scale ----------
// block = (seg, h, b); 256 threads, 8 waves, each wave owns 32 queries.
#define ATTN_SMEM_BYTES ((3 * 256 * 72 + 8 * 32 * 34) * 2)  // 128000 B

__global__ __launch_bounds__(256) void attn_kernel(const unsigned short* __restrict__ Qg,
                                                   const unsigned short* __restrict__ Kg,
                                                   const unsigned short* __restrict__ Vg,
                                                   float* __restrict__ macc,
                                                   float* __restrict__ zacc,
                                                   float* __restrict__ oacc,
                                                   int w, int r) {
  extern __shared__ __align__(16) unsigned short sm[];
  unsigned short* Qs = sm;                  // [256][72]
  unsigned short* Ks = sm + 256 * 72;
  unsigned short* Vs = sm + 2 * 256 * 72;
  unsigned short* Ps = sm + 3 * 256 * 72;   // [8][32][34]

  const int tid = threadIdx.x, lane = tid & 31, wid = tid >> 5;
  const int lh = lane >> 4, n16 = lane & 15;
  const int seg = blockIdx.x, h = blockIdx.y, b = blockIdx.z;
  const int g = h / (H_ / r);
  const int segBase = seg * w;

  const size_t headBase = ((size_t)b * H_ + h) * T_ * D_;
  const unsigned short* Qh = Qg + headBase;
  const unsigned short* Kh = Kg + headBase;
  const unsigned short* Vh = Vg + headBase;

  // async-DMA gather of 256 rows (stride r, offset g) of Q,K,V into LDS
#pragma unroll
  for (int i = 0; i < 8; ++i) {
    int idx = tid + 256 * i;             // 2048 chunks of 8 bf16
    int row = idx >> 3, c8 = (idx & 7) * 8;
    int t = segBase + row * r + g;
    unsigned voff = (unsigned)((((size_t)t) * D_ + c8) * 2);
    async_g2l_b128(Qh, voff, lds_off(&Qs[row * 72 + c8]));
    async_g2l_b128(Kh, voff, lds_off(&Ks[row * 72 + c8]));
    async_g2l_b128(Vh, voff, lds_off(&Vs[row * 72 + c8]));
  }
  wait_async0();
  __syncthreads();

  const int qb = wid * 32;
  unsigned short* Pw = Ps + wid * (32 * 34);

  // Q A-fragments (2 M-tiles x 2 K-chunks of 32)
  v16bf aq[2][2];
#pragma unroll
  for (int mi = 0; mi < 2; ++mi)
#pragma unroll
    for (int c = 0; c < 2; ++c) {
      v8ui u;
      int ro = (qb + mi * 16 + n16) * 72 + c * 32;
#pragma unroll
      for (int v = 0; v < 8; ++v)
        u[v] = *(const unsigned int*)&Qs[ro + a_base(v) + 8 * lh];
      aq[mi][c] = bf_from_pairs(u);
    }

  v8f zero = {0.f, 0.f, 0.f, 0.f, 0.f, 0.f, 0.f, 0.f};
  v8f O[2][4];
  float mrow[2][8], lrow[2][8];
#pragma unroll
  for (int mi = 0; mi < 2; ++mi) {
#pragma unroll
    for (int nt = 0; nt < 4; ++nt) O[mi][nt] = zero;
#pragma unroll
    for (int v = 0; v < 8; ++v) { mrow[mi][v] = -1e30f; lrow[mi][v] = 0.f; }
  }

  // online flash over key chunks of 32 (causal: only chunks <= wid)
  for (int jj = 0; jj <= wid; ++jj) {
    const int kb = jj * 32;
    v8f st[2][2];
#pragma unroll
    for (int t16 = 0; t16 < 2; ++t16) {
      int kb16 = kb + t16 * 16;
      v16bf bk[2];
#pragma unroll
      for (int c = 0; c < 2; ++c) {     // B = K^T: lane = key, elem = dd (table order)
        v16us u;
#pragma unroll
        for (int e = 0; e < 16; ++e)
          u[e] = Ks[(kb16 + n16) * 72 + c * 32 + a_base(e >> 1) + (e & 1) + 8 * lh];
        bk[c] = bf_from_us(u);
      }
#pragma unroll
      for (int mi = 0; mi < 2; ++mi) {
        v8f s = zero;
        s = wmma_bf16(aq[mi][0], bk[0], s);
        s = wmma_bf16(aq[mi][1], bk[1], s);
        st[mi][t16] = s;
      }
    }

    const int kj0 = kb + n16, kj1 = kb + 16 + n16;
#pragma unroll
    for (int mi = 0; mi < 2; ++mi)
#pragma unroll
      for (int v = 0; v < 8; ++v) {
        int qi = qb + mi * 16 + v + 8 * lh;
        float s0 = (kj0 <= qi) ? st[mi][0][v] : -1e30f;
        float s1 = (kj1 <= qi) ? st[mi][1][v] : -1e30f;
        float mx = fmaxf(s0, s1);
#pragma unroll
        for (int mk = 1; mk < 16; mk <<= 1)
          mx = fmaxf(mx, __shfl_xor(mx, mk, 32));
        float mnew = fmaxf(mrow[mi][v], mx);
        float sc = __expf(mrow[mi][v] - mnew);
        mrow[mi][v] = mnew;
        lrow[mi][v] *= sc;
#pragma unroll
        for (int nt = 0; nt < 4; ++nt) O[mi][nt][v] *= sc;
        float p0 = __expf(s0 - mnew);
        float p1 = __expf(s1 - mnew);
        lrow[mi][v] += p0 + p1;
        int prow = mi * 16 + v + 8 * lh;
        Pw[prow * 34 + n16] = f2bf(p0);
        Pw[prow * 34 + 16 + n16] = f2bf(p1);
      }

    // PV: P (32q x 32k) as A, V chunk (32k x 64) as B
#pragma unroll
    for (int mi = 0; mi < 2; ++mi) {
      v8ui pu;
      int ro = (mi * 16 + n16) * 34;
#pragma unroll
      for (int v = 0; v < 8; ++v)
        pu[v] = *(const unsigned int*)&Pw[ro + a_base(v) + 8 * lh];
      v16bf ap = bf_from_pairs(pu);
#pragma unroll
      for (int nt = 0; nt < 4; ++nt) {
        v8ui vu;
        int co = kb * 72 + nt * 16 + n16;
#pragma unroll
        for (int j = 0; j < 8; ++j)
          vu[j] = (unsigned)Vs[co + (a_base(j) + 8 * lh) * 72] |
                  ((unsigned)Vs[co + (a_base(j) + 8 * lh + 1) * 72] << 16);
        O[mi][nt] = wmma_bf16(ap, bf_from_pairs(vu), O[mi][nt]);
      }
    }
  }

  // finalize + streaming LSE merge into (macc, zacc, oacc)
#pragma unroll
  for (int mi = 0; mi < 2; ++mi)
#pragma unroll
    for (int v = 0; v < 8; ++v) {
      float l = lrow[mi][v];
#pragma unroll
      for (int mk = 1; mk < 16; mk <<= 1) l += __shfl_xor(l, mk, 32);
      float lse = mrow[mi][v] + __logf(l);
      float inv = 1.0f / l;
      int qi = qb + mi * 16 + v + 8 * lh;
      int t = segBase + qi * r + g;
      size_t midx = ((size_t)b * T_ + t) * H_ + h;
      float mold = macc[midx];
      float mn = fmaxf(mold, lse);
      float so = __expf(mold - mn);
      float wg = __expf(lse - mn);
      size_t ob = midx * D_;
#pragma unroll
      for (int nt = 0; nt < 4; ++nt) {
        int col = nt * 16 + n16;
        oacc[ob + col] = oacc[ob + col] * so + wg * (O[mi][nt][v] * inv);
      }
      if (n16 == 0) {
        zacc[midx] = zacc[midx] * so + wg;
        macc[midx] = mn;
      }
    }
}

// ---------- host orchestration ----------
extern "C" void kernel_launch(void* const* d_in, const int* in_sizes, int n_in,
                              void* d_out, int out_size, void* d_ws, size_t ws_size,
                              hipStream_t stream) {
  (void)in_sizes; (void)n_in; (void)out_size; (void)ws_size;

  const float* x   = (const float*)d_in[0];
  const float* wq  = (const float*)d_in[1];
  const float* bq  = (const float*)d_in[2];
  const float* wk  = (const float*)d_in[3];
  const float* bk  = (const float*)d_in[4];
  const float* wvw = (const float*)d_in[5];
  const float* bvv = (const float*)d_in[6];
  const float* wo  = (const float*)d_in[7];
  const float* bo  = (const float*)d_in[8];

  char* p = (char*)d_ws;
  auto alloc = [&](size_t bytes) {
    void* r = (void*)p;
    p += (bytes + 255) & ~(size_t)255;
    return r;
  };

  const size_t nX = (size_t)MROWS * E_;          // 8388608
  const size_t nW = (size_t)E_ * E_;             // 1048576
  const size_t nBTH = (size_t)B_ * T_ * H_;      // 131072

  unsigned short* xbf = (unsigned short*)alloc(nX * 2);
  unsigned short* wqb = (unsigned short*)alloc(nW * 2);
  unsigned short* wkb = (unsigned short*)alloc(nW * 2);
  unsigned short* wvb = (unsigned short*)alloc(nW * 2);
  unsigned short* wob = (unsigned short*)alloc(nW * 2);
  unsigned short* Qb  = (unsigned short*)alloc(nX * 2);
  unsigned short* Kb  = (unsigned short*)alloc(nX * 2);
  unsigned short* Vb  = (unsigned short*)alloc(nX * 2);
  float* macc = (float*)alloc(nBTH * 4);
  float* zacc = (float*)alloc(nBTH * 4);
  float* oacc = (float*)alloc(nBTH * (size_t)D_ * 4);
  unsigned short* Abf = (unsigned short*)alloc(nX * 2);

  // init merge state (must re-init every call: deterministic replays)
  hipMemsetAsync(zacc, 0, nBTH * 4, stream);
  hipMemsetAsync(oacc, 0, nBTH * (size_t)D_ * 4, stream);
  fill_kernel<<<512, 256, 0, stream>>>(macc, -1e30f, (int)nBTH);

  // f32 -> bf16 conversions
  cvt_bf16_kernel<<<4096, 256, 0, stream>>>(x,   xbf, (int)nX);
  cvt_bf16_kernel<<<2048, 256, 0, stream>>>(wq,  wqb, (int)nW);
  cvt_bf16_kernel<<<2048, 256, 0, stream>>>(wk,  wkb, (int)nW);
  cvt_bf16_kernel<<<2048, 256, 0, stream>>>(wvw, wvb, (int)nW);
  cvt_bf16_kernel<<<2048, 256, 0, stream>>>(wo,  wob, (int)nW);

  // QKV projections (Q scaled by d^-0.5 = 0.125)
  dim3 ggrid(E_ / 64, MROWS / 128);
  gemm_kernel<<<ggrid, 256, 0, stream>>>(xbf, wqb, bq, 0.125f, 1, Qb, nullptr);
  gemm_kernel<<<ggrid, 256, 0, stream>>>(xbf, wkb, bk, 1.0f,   1, Kb, nullptr);
  gemm_kernel<<<ggrid, 256, 0, stream>>>(xbf, wvb, bvv, 1.0f,  1, Vb, nullptr);

  // attention scales, sequential on stream (race-free streaming merge)
  hipFuncSetAttribute(reinterpret_cast<const void*>(attn_kernel),
                      hipFuncAttributeMaxDynamicSharedMemorySize, ATTN_SMEM_BYTES);
  const int wsegs[5] = {256, 512, 1024, 2048, 4096};
  const int rdils[5] = {1, 2, 4, 8, 16};
  for (int i = 0; i < 5; ++i) {
    dim3 agrid(T_ / wsegs[i], H_, B_);
    attn_kernel<<<agrid, 256, ATTN_SMEM_BYTES, stream>>>(Qb, Kb, Vb, macc, zacc,
                                                         oacc, wsegs[i], rdils[i]);
  }

  // merge -> bf16 [B*T, E]
  merge_kernel<<<4096, 256, 0, stream>>>(oacc, zacc, Abf, (int)nX);

  // final projection -> f32 d_out
  gemm_kernel<<<ggrid, 256, 0, stream>>>(Abf, wob, bo, 1.0f, 0, nullptr,
                                         (float*)d_out);
}
